// Kernel_4005987118033986640_53472342835842
// MI455X (gfx1250) — matrix operand order repaired and checked by reading, not compiled, not run on hardware
//
#include <hip/hip_runtime.h>
#include <hip/hip_bf16.h>
#include <math.h>

// ---------------------------------------------------------------------------
// Problem constants (from reference)
// ---------------------------------------------------------------------------
#define Cc   128
#define Hh   64
#define Ww   64
#define Rr   7
#define Kk   3
#define HW   (Hh * Ww)          // 4096
#define M1   (Cc * Rr * Kk)     // 2688  (c*21)
#define NB   4                  // batch
// Padded x copy: x[c][i][j] lives at xpad[c][i+9][j+8]; tap (r,k) of pixel
// (i,j) is xpad[c][i+3r][j+2k+6].  WP=80 (=160B rows) keeps rows 16B-aligned.
#define HP   82
#define WP   80

typedef __attribute__((ext_vector_type(16))) __bf16 v16bf;
typedef __attribute__((ext_vector_type(8)))  float  v8f;

union FragB { v16bf v; uint4 q[2]; };
union FragA { v16bf v; unsigned int u[8]; };

// ---------------------------------------------------------------------------
// Phase 0: build zero-padded bf16 copy of x. Removes all per-iteration
// cndmask padding selects and f32->bf16 converts from the GEMM hot loops.
// ---------------------------------------------------------------------------
__global__ __launch_bounds__(256)
void build_xpad(const float* __restrict__ x, __bf16* __restrict__ xpad) {
    size_t idx = (size_t)blockIdx.x * 256 + threadIdx.x;
    if (idx >= (size_t)NB * Cc * HP * WP) return;
    int wp   = (int)(idx % WP);
    size_t r = idx / WP;
    int hp   = (int)(r % HP);
    size_t nc = r / HP;
    int i = hp - 9, j = wp - 8;
    float v = 0.0f;
    if (i >= 0 && i < Hh && j >= 0 && j < Ww)
        v = x[nc * HW + (size_t)i * Ww + j];
    xpad[idx] = (__bf16)v;
}

// ---------------------------------------------------------------------------
// Phase 1: per (n, rk):  T3(c,d) = Xshift(r,k) @ X^T   (128 x 128, K = 4096)
//          t6t[n][d][c*21+rk] = p6_w[c,rk,d] * T3(c,d) / 64    (bf16, d-major)
// Block = 8 waves; wave w owns c-rows [16w,16w+16) x all 128 d (8 accumulators,
// 8 WMMAs per K-chunk).  B chunk (128 d x 32 K) staged in LDS.
// ---------------------------------------------------------------------------
__global__ __launch_bounds__(256)
void t6_gemm_bf16_wmma(const __bf16* __restrict__ xpad,
                       const float* __restrict__ p6w,
                       __bf16* __restrict__ t6t,
                       float inv_sqrt_hw) {
    __shared__ __bf16 Bs[Cc * 32];                  // [d][K] 8 KB

    const int blk  = blockIdx.x;                    // n*21 + rk
    const int n    = blk / (Rr * Kk);
    const int rk   = blk % (Rr * Kk);
    const int r    = rk / Kk;
    const int k    = rk % Kk;
    const int wave = threadIdx.x >> 5;
    const int lane = threadIdx.x & 31;
    const int lo   = lane & 15;
    const int hi   = lane >> 4;

    const __bf16* xp = xpad + (size_t)n * Cc * HP * WP;
    const int cA = wave * 16 + lo;                  // A row (c) for this lane

    v8f acc[8];
#pragma unroll
    for (int t = 0; t < 8; ++t)
        acc[t] = (v8f){0.f, 0.f, 0.f, 0.f, 0.f, 0.f, 0.f, 0.f};

    // cooperative-fill mapping: thread -> (d = tid>>1, 16 cols at (tid&1)*16)
    const int dF   = threadIdx.x >> 1;
    const int colF = (threadIdx.x & 1) * 16;

    for (int kb = 0; kb < HW; kb += 32) {
        const int i  = kb >> 6;                      // source image row
        const int j0 = kb & 63;                      // 0 or 32

        // ---- stage B chunk: Bs[d][0..31] = x[n][d][kb..kb+31] (identity tap)
        {
            const uint4* src = (const uint4*)(xp + (size_t)dF * HP * WP
                                              + (size_t)(i + 9) * WP + (j0 + colF) + 8);
            uint4* dst = (uint4*)(Bs + dF * 32 + colF);
            dst[0] = src[0];                         // ds_store_b128, 16B aligned
            dst[1] = src[1];
        }
        __syncthreads();

        // ---- A fragment: tap (r,k) of rows cA, dword loads (4B aligned)
        FragA A;
        {
            const unsigned int* arow32 =
                (const unsigned int*)(xp + (size_t)cA * HP * WP
                                      + (size_t)(i + 3 * r) * WP + j0 + 2 * k + 6);
            const int base = hi ? 4 : 0;
#pragma unroll
            for (int q = 0; q < 8; ++q)              // ISA 16-bit A 16x32 layout
                A.u[q] = arow32[base + q + (q >= 4 ? 4 : 0)];
        }

        // ---- 8 WMMAs sweeping all 128 d
#pragma unroll
        for (int nt = 0; nt < 8; ++nt) {
            FragB B;
            const uint4* bp = (const uint4*)(Bs + (nt * 16 + lo) * 32 + (hi ? 8 : 0));
            B.q[0] = bp[0];                          // ds_load_b128 x2
            B.q[1] = bp[2];
            acc[nt] = __builtin_amdgcn_wmma_f32_16x16x32_bf16(
                          false, A.v, false, B.v, (short)0, acc[nt], false, false);
        }
        __syncthreads();
    }

    // ---- epilogue: t6 = p6_w * t3 / 64, stored transposed (d-major) as bf16
#pragma unroll
    for (int nt = 0; nt < 8; ++nt) {
        const int d = nt * 16 + lo;
        __bf16* dst = t6t + ((size_t)n * Cc + d) * M1;
#pragma unroll
        for (int v = 0; v < 8; ++v) {
            const int c  = wave * 16 + v + 8 * hi;   // C/D layout: M = v + 8*hi
            const int gm = c * (Rr * Kk) + rk;
            dst[gm] = (__bf16)(acc[nt][v] * inv_sqrt_hw * p6w[(size_t)gm * Cc + d]);
        }
    }
}

// ---------------------------------------------------------------------------
// Phase 2:  out[n,d,hw] = sum_kc (1+p5_w[kc,hw]) * t2[n,kc,hw] * t6[n,kc,d] / sqrt(2688)
// GEMM: M = hw (4096), N = d (128), K = kc (2688).
// Block covers 32 hw x 128 d; 8 waves = 2 hw-strips x 4 d-pairs (2 WMMA each).
// t7 chunk (32 hw x 32 kc) built cooperatively in LDS (hw-major) from
// coalesced p5_w float4 reads + xpad dword reads.
// ---------------------------------------------------------------------------
__global__ __launch_bounds__(256)
void out_gemm_bf16_wmma(const __bf16* __restrict__ xpad,
                        const float* __restrict__ p5w,
                        const __bf16* __restrict__ t6t,
                        float* __restrict__ out,
                        float inv_sqrt_ck) {
    __shared__ __bf16 As[32 * 32];                  // [hw_local][kc_local] 2 KB

    const int blk  = blockIdx.x;                    // n * 128 tiles of 32 hw
    const int n    = blk >> 7;
    const int mt   = blk & 127;
    const int hw0  = mt * 32;
    const int wave = threadIdx.x >> 5;
    const int lane = threadIdx.x & 31;
    const int lo   = lane & 15;
    const int hi   = lane >> 4;
    const int mrow  = wave & 1;                     // 16-hw strip
    const int dpair = wave >> 1;                    // 2 d-tiles of 16

    const __bf16* xp = xpad + (size_t)n * Cc * HP * WP;

    v8f accs[2];
    accs[0] = (v8f){0.f, 0.f, 0.f, 0.f, 0.f, 0.f, 0.f, 0.f};
    accs[1] = accs[0];

    // fill mapping: thread -> (kc row = tid>>3, 4 contiguous hw at (tid&7)*4)
    const int kcrF = threadIdx.x >> 3;
    const int hwqF = (threadIdx.x & 7) * 4;

    for (int kb = 0; kb < M1; kb += 32) {
        // ---- stage t7 chunk (hw-major so A fragments are contiguous in LDS)
        {
            const int kc = kb + kcrF;
            const int c  = kc / (Rr * Kk);
            const int rr = kc % (Rr * Kk);
            const int r  = rr / Kk;
            const int kk = rr % Kk;
            const int hwb = hw0 + hwqF;
            const int i = hwb >> 6, j = hwb & 63;

            if (kb + 32 < M1)                        // prefetch dominant p5_w stream
                __builtin_prefetch(&p5w[(size_t)(kb + 32 + kcrF) * HW + hwb], 0, 1);

            const float4 w = *(const float4*)(p5w + (size_t)kc * HW + hwb);
            const unsigned int* xr =
                (const unsigned int*)(xp + (size_t)c * HP * WP
                                      + (size_t)(i + 3 * r) * WP + j + 2 * kk + 6);
            union { unsigned int u; __bf16 h[2]; } x0, x1;
            x0.u = xr[0];  x1.u = xr[1];
            As[(hwqF + 0) * 32 + kcrF] = (__bf16)((1.0f + w.x) * (float)x0.h[0]);
            As[(hwqF + 1) * 32 + kcrF] = (__bf16)((1.0f + w.y) * (float)x0.h[1]);
            As[(hwqF + 2) * 32 + kcrF] = (__bf16)((1.0f + w.z) * (float)x1.h[0]);
            As[(hwqF + 3) * 32 + kcrF] = (__bf16)((1.0f + w.w) * (float)x1.h[1]);
        }
        __syncthreads();

        // ---- A fragment from LDS: row hw_local = mrow*16+lo, aligned b128 x2
        FragB A;                                      // (same 2x uint4 shape)
        {
            const __bf16* arow = As + (mrow * 16 + lo) * 32;
            A.q[0] = *(const uint4*)(arow + (hi ? 16 : 0));
            A.q[1] = *(const uint4*)(arow + 8 + (hi ? 16 : 0));
        }

        // ---- B fragments straight from transposed bf16 t6 (global b128 x2)
#pragma unroll
        for (int s = 0; s < 2; ++s) {
            const int d = (dpair * 2 + s) * 16 + lo;
            FragB B;
            const uint4* bp = (const uint4*)(t6t + ((size_t)n * Cc + d) * M1
                                             + kb + (hi ? 16 : 0));
            B.q[0] = bp[0];
            B.q[1] = bp[1];
            accs[s] = __builtin_amdgcn_wmma_f32_16x16x32_bf16(
                          false, A.v, false, B.v, (short)0, accs[s], false, false);
        }
        __syncthreads();
    }

    // ---- epilogue: aligned float4 stores, out layout (n, d, hw)
#pragma unroll
    for (int s = 0; s < 2; ++s) {
        const int d = (dpair * 2 + s) * 16 + lo;
        float* orow = out + ((size_t)n * Cc + d) * HW + hw0 + mrow * 16 + hi * 8;
        const v8f a = accs[s];
        float4 f0 = {a[0] * inv_sqrt_ck, a[1] * inv_sqrt_ck,
                     a[2] * inv_sqrt_ck, a[3] * inv_sqrt_ck};
        float4 f1 = {a[4] * inv_sqrt_ck, a[5] * inv_sqrt_ck,
                     a[6] * inv_sqrt_ck, a[7] * inv_sqrt_ck};
        *(float4*)(orow)     = f0;
        *(float4*)(orow + 4) = f1;
    }
}

// ---------------------------------------------------------------------------
// Host-side launcher
// ---------------------------------------------------------------------------
extern "C" void kernel_launch(void* const* d_in, const int* in_sizes, int n_in,
                              void* d_out, int out_size, void* d_ws, size_t ws_size,
                              hipStream_t stream) {
    const float* x   = (const float*)d_in[0];   // (4,128,64,64)
    const float* p5w = (const float*)d_in[1];   // (1,128,7,3,64,64)
    const float* p6w = (const float*)d_in[2];   // (1,128,7,3,128)
    float*       out = (float*)d_out;           // (4,128,64,64)

    // workspace layout: xpad (6.7 MB bf16) | t6t (2.75 MB bf16, d-major)
    __bf16* xpad = (__bf16*)d_ws;
    __bf16* t6t  = xpad + (size_t)NB * Cc * HP * WP;

    const float inv_sqrt_hw = 1.0f / sqrtf((float)HW);   // 1/64
    const float inv_sqrt_ck = 1.0f / sqrtf((float)M1);   // 1/sqrt(2688)

    const int padN = (int)(((size_t)NB * Cc * HP * WP + 255) / 256);
    build_xpad<<<padN, 256, 0, stream>>>(x, xpad);

    // Phase 1: one block per (n, rk) shift — 84 blocks of 8 waves, 8 WMMAs/chunk
    t6_gemm_bf16_wmma<<<NB * Rr * Kk, 256, 0, stream>>>(xpad, p6w, t6t, inv_sqrt_hw);

    // Phase 2: 4 * 128 blocks of 8 waves, 2 WMMAs/chunk/wave
    out_gemm_bf16_wmma<<<NB * (HW / 32), 256, 0, stream>>>(xpad, p5w, t6t, out, inv_sqrt_ck);
}
